// BN_26654567039559
// MI455X (gfx1250) — matrix operand order repaired and checked by reading, not compiled, not run on hardware
//
#include <hip/hip_runtime.h>

// MI455X / gfx1250: wave32, WMMA 16x16x32 f16 -> f32 accumulate.
// Layer-1 B operand (80KB, > 64KB WGP$) staged into LDS via TDM
// (tensor_load_to_lds + s_wait_tensorcnt) once per workgroup.

typedef __attribute__((ext_vector_type(16))) _Float16 v16h;
typedef __attribute__((ext_vector_type(8)))  _Float16 v8h;
typedef __attribute__((ext_vector_type(8)))  float    v8f;
typedef __attribute__((ext_vector_type(4)))  unsigned int u32x4;
typedef __attribute__((ext_vector_type(8)))  int i32x8;
typedef __attribute__((ext_vector_type(4)))  int i32x4;

#define N_NODES 100000
#define N_EDGES 1000000

#if __has_builtin(__builtin_amdgcn_tensor_load_to_lds) && \
    __has_builtin(__builtin_amdgcn_s_wait_tensorcnt)
#define USE_TDM 1
#else
#define USE_TDM 0
#endif

static __device__ __forceinline__ v8f wmma16(v16h a, v16h b, v8f c) {
  return __builtin_amdgcn_wmma_f32_16x16x32_f16(false, a, false, b, (short)0, c,
                                                false, false);
}

static __device__ __forceinline__ v8f vzero8() {
  v8f z = {0.f, 0.f, 0.f, 0.f, 0.f, 0.f, 0.f, 0.f};
  return z;
}

static __device__ __forceinline__ v8h hzero8() {
  v8h z = {(_Float16)0.f, (_Float16)0.f, (_Float16)0.f, (_Float16)0.f,
           (_Float16)0.f, (_Float16)0.f, (_Float16)0.f, (_Float16)0.f};
  return z;
}

#if USE_TDM
// Issue one TDM DMA: copy `nwords` dwords from global `gptr` to LDS offset
// `lds_off`. D# built per CDNA5 ISA 8.3/8.4 (count=1, type=2 "image",
// data_size=4B, single row tile: tile_dim0 = nwords (<=65535), tile_dim1=1).
// Caller must be a single wave; follow with s_wait_tensorcnt + barrier.
// This toolchain's builtin is the 6-arg form:
//   (u32x4 g0, i32x8 g1, i32x4 g2, i32x4 g3, i32x8 extra, i32 cpol)
static __device__ __forceinline__ void tdm_load_to_lds(unsigned lds_off,
                                                       const void* gptr,
                                                       unsigned nwords) {
  unsigned long long ga = (unsigned long long)gptr;
  u32x4 g0;
  g0[0] = 1u;                                        // count=1 (valid user D#)
  g0[1] = lds_off;                                   // lds_addr [63:32]
  g0[2] = (unsigned)(ga & 0xFFFFFFFFu);              // global_addr[31:0]
  g0[3] = (unsigned)((ga >> 32) & 0x01FFFFFFu)       // global_addr[56:32]
          | (2u << 30);                              // type=2
  i32x8 g1;
  g1[0] = (int)(2u << 16);                           // data_size=4B, mask=0
  g1[1] = (int)((nwords & 0xFFFFu) << 16);           // tensor_dim0[15:0] @ [63:48]
  g1[2] = (int)((nwords >> 16) & 0xFFFFu)            // tensor_dim0[31:16] @ [79:64]
          | (int)(1u << 16);                         // tensor_dim1[15:0]=1 @ [95:80]
  g1[3] = (int)((nwords & 0xFFFFu) << 16);           // tile_dim0 @ [127:112]
  g1[4] = 1;                                         // tile_dim1=1, tile_dim2=0
  g1[5] = (int)nwords;                               // tensor_dim0_stride[31:0]
  g1[6] = 0;                                         // stride hi / dim1_stride lo
  g1[7] = 0;
  i32x4 z4 = {0, 0, 0, 0};                           // groups 2/3 unused (<=2D)
  i32x8 z8 = {0, 0, 0, 0, 0, 0, 0, 0};
  __builtin_amdgcn_tensor_load_to_lds(g0, g1, z4, z4, z8, 0);
}
#endif

// ---------------------------------------------------------------------------
// Pack f32 W[K x N] (row-major) into WMMA B-operand layout:
//   K = kt*32 + (j>=8 ? 16 : 0) + (l>=16 ? 8 : 0) + (j & 7),  N = nt*16 + (l & 15)
// so each lane loads one v16h (32B) per (kt,nt).
// ---------------------------------------------------------------------------
__global__ void pack_b_kernel(const float* __restrict__ W, _Float16* __restrict__ out,
                              int Kvalid, int N, int KT) {
  int t = blockIdx.x * blockDim.x + threadIdx.x;
  int NT = N >> 4;
  int total = KT * NT * 512;
  if (t >= total) return;
  int j = t & 15;
  int lane = (t >> 4) & 31;
  int tile = t >> 9;
  int nt = tile % NT;
  int kt = tile / NT;
  int k = kt * 32 + ((j >> 3) << 4) + ((lane >> 4) << 3) + (j & 7);
  int n = nt * 16 + (lane & 15);
  out[t] = (k < Kvalid) ? (_Float16)W[(long)k * N + n] : (_Float16)0.f;
}

// ---------------------------------------------------------------------------
__global__ void node_encoder_kernel(const float* __restrict__ nf,
                                    const float* __restrict__ nw,
                                    const float* __restrict__ nb,
                                    float* __restrict__ x,
                                    _Float16* __restrict__ xh) {
  int t = blockIdx.x * blockDim.x + threadIdx.x;
  if (t >= N_NODES) return;
  float f0 = nf[t * 3 + 0], f1 = nf[t * 3 + 1], f2 = nf[t * 3 + 2];
#pragma unroll
  for (int j = 0; j < 16; ++j) {
    float s = nb[j] + f0 * nw[j] + f1 * nw[16 + j] + f2 * nw[32 + j];
    s = s > 0.f ? s : 0.f;
    x[(long)t * 32 + j] = s;
    xh[(long)t * 32 + j] = (_Float16)s;
  }
}

// ---------------------------------------------------------------------------
// Fused edge encoder. One wave = 16 edges, 4 waves/block.
// Dynamic LDS layout (bytes):
//   [0, 81920)            w1s  : packed layer-1 weights (TDM-staged)
//   [81920, 98304)        lds1 : per-wave 16x128 f16 relayout tile
//   [98304, 106496)       lds2 : per-wave 16x64 f16 relayout tile
// ---------------------------------------------------------------------------
#define W1_HALVES (10 * 8 * 512)  // 40960 halves = 81920 B

__global__ __launch_bounds__(128)
void edge_encoder_kernel(const float* __restrict__ ef,
                         const _Float16* __restrict__ w1p, const float* __restrict__ b1,
                         const _Float16* __restrict__ w2p, const float* __restrict__ b2,
                         const _Float16* __restrict__ w3p, const float* __restrict__ b3,
                         _Float16* __restrict__ eh) {
  extern __shared__ _Float16 smem[];
  _Float16* w1s = smem;
  const int lane = threadIdx.x & 31;
  const int w    = threadIdx.x >> 5;
  _Float16* lds1 = smem + W1_HALVES + w * (16 * 128);
  _Float16* lds2 = smem + W1_HALVES + 4 * (16 * 128) + w * (16 * 64);

  // ---- stage layer-1 B operand into LDS ----
#if USE_TDM
  if (threadIdx.x < 32) {  // wave 0 issues one TDM DMA (EXEC-independent)
    tdm_load_to_lds((unsigned)(unsigned long long)(void*)w1s, w1p,
                    W1_HALVES / 2 /* dwords */);
    __builtin_amdgcn_s_wait_tensorcnt(0);
  }
#else
  for (int c = threadIdx.x; c < W1_HALVES / 8; c += 128)
    ((v8h*)w1s)[c] = ((const v8h*)w1p)[c];
#endif
  __syncthreads();

  const long r0  = ((long)blockIdx.x * 4 + w) * 16;
  const int r     = lane & 15;
  const int half8 = (lane >> 4) * 8;
  const int mrow  = (lane >> 4) * 8;  // D-layout row base

  // ----- layer 1: [16x301] @ [301x128] (K padded to 320), B from LDS -----
  v8f acc[8];
#pragma unroll
  for (int nt = 0; nt < 8; ++nt) acc[nt] = vzero8();

  const float* efr = ef + (r0 + r) * 301;
  for (int kt = 0; kt < 10; ++kt) {
    v16h a;
    int kb = kt * 32 + half8;
#pragma unroll
    for (int j = 0; j < 8; ++j) {
      int k = kb + j;
      a[j] = (k < 301) ? (_Float16)efr[k] : (_Float16)0.f;
    }
#pragma unroll
    for (int j = 0; j < 8; ++j) {
      int k = kb + 16 + j;
      a[8 + j] = (k < 301) ? (_Float16)efr[k] : (_Float16)0.f;
    }
    const _Float16* wp = w1s + kt * 8 * 512;
#pragma unroll
    for (int nt = 0; nt < 8; ++nt) {
      v16h b = *(const v16h*)(wp + (nt * 32 + lane) * 16);
      acc[nt] = wmma16(a, b, acc[nt]);
    }
  }
#pragma unroll
  for (int nt = 0; nt < 8; ++nt) {
    int col = nt * 16 + r;
    float bias = b1[col];
#pragma unroll
    for (int i = 0; i < 8; ++i) {
      float v = acc[nt][i] + bias;
      lds1[(mrow + i) * 128 + col] = (_Float16)(v > 0.f ? v : 0.f);
    }
  }
  __syncthreads();

  // ----- layer 2: [16x128] @ [128x64] -----
  v8f acc2[4];
#pragma unroll
  for (int nt = 0; nt < 4; ++nt) acc2[nt] = vzero8();
  const _Float16* L1 = lds1 + r * 128;
#pragma unroll
  for (int kt = 0; kt < 4; ++kt) {
    v16h a;
    int kb = kt * 32 + half8;
#pragma unroll
    for (int j = 0; j < 8; ++j) a[j] = L1[kb + j];
#pragma unroll
    for (int j = 0; j < 8; ++j) a[8 + j] = L1[kb + 16 + j];
    const _Float16* wp = w2p + (long)kt * 4 * 512;
#pragma unroll
    for (int nt = 0; nt < 4; ++nt) {
      v16h b = *(const v16h*)(wp + (nt * 32 + lane) * 16);
      acc2[nt] = wmma16(a, b, acc2[nt]);
    }
  }
#pragma unroll
  for (int nt = 0; nt < 4; ++nt) {
    int col = nt * 16 + r;
    float bias = b2[col];
#pragma unroll
    for (int i = 0; i < 8; ++i) {
      float v = acc2[nt][i] + bias;
      lds2[(mrow + i) * 64 + col] = (_Float16)(v > 0.f ? v : 0.f);
    }
  }
  __syncthreads();

  // ----- layer 3: [16x64] @ [64x32] -----
  v8f acc3[2];
  acc3[0] = vzero8();
  acc3[1] = vzero8();
  const _Float16* L2 = lds2 + r * 64;
#pragma unroll
  for (int kt = 0; kt < 2; ++kt) {
    v16h a;
    int kb = kt * 32 + half8;
#pragma unroll
    for (int j = 0; j < 8; ++j) a[j] = L2[kb + j];
#pragma unroll
    for (int j = 0; j < 8; ++j) a[8 + j] = L2[kb + 16 + j];
    const _Float16* wp = w3p + (long)kt * 2 * 512;
#pragma unroll
    for (int nt = 0; nt < 2; ++nt) {
      v16h b = *(const v16h*)(wp + (nt * 32 + lane) * 16);
      acc3[nt] = wmma16(a, b, acc3[nt]);
    }
  }
#pragma unroll
  for (int nt = 0; nt < 2; ++nt) {
    int col = nt * 16 + r;
    float bias = b3[col];
#pragma unroll
    for (int i = 0; i < 8; ++i) {
      float v = acc3[nt][i] + bias;
      eh[(r0 + mrow + i) * 32 + col] = (_Float16)(v > 0.f ? v : 0.f);
    }
  }
}

// ---------------------------------------------------------------------------
// NNConv messages: msg = concat(x[src], e) @ net_w + net_b ; atomic scatter.
// A 8-half runs never straddle the x/e boundary -> one 16B load per run.
// ---------------------------------------------------------------------------
template <int FIN>
static __device__ __forceinline__ v8h msg_run(const _Float16* __restrict__ xr,
                                              const _Float16* __restrict__ er,
                                              int kb) {
  if (kb < FIN) return *(const v8h*)(xr + kb);
  if (kb < FIN + 32) return *(const v8h*)(er + (kb - FIN));
  return hzero8();
}

template <int FIN>
__global__ __launch_bounds__(128)
void nnconv_msg_kernel(const _Float16* __restrict__ xh,
                       const _Float16* __restrict__ eh,
                       const int* __restrict__ src,
                       const int* __restrict__ dst,
                       const _Float16* __restrict__ wnp,
                       const float* __restrict__ nb,
                       float* __restrict__ agg) {
  const int lane = threadIdx.x & 31;
  const int w    = threadIdx.x >> 5;
  const long r0  = ((long)blockIdx.x * 4 + w) * 16;
  const int r     = lane & 15;
  const int half8 = (lane >> 4) * 8;

  const long rowE = r0 + r;
  const int s = src[rowE];
  const _Float16* xr = xh + (long)s * 32;
  const _Float16* er = eh + rowE * 32;

  v8f acc[2];
  acc[0] = vzero8();
  acc[1] = vzero8();
#pragma unroll
  for (int kt = 0; kt < 2; ++kt) {
    int kb = kt * 32 + half8;
    v8h lo = msg_run<FIN>(xr, er, kb);
    v8h hi = msg_run<FIN>(xr, er, kb + 16);
    v16h a;
#pragma unroll
    for (int j = 0; j < 8; ++j) { a[j] = lo[j]; a[8 + j] = hi[j]; }
    const _Float16* wp = wnp + (long)kt * 2 * 512;
#pragma unroll
    for (int nt = 0; nt < 2; ++nt) {
      v16h b = *(const v16h*)(wp + (nt * 32 + lane) * 16);
      acc[nt] = wmma16(a, b, acc[nt]);
    }
  }
#pragma unroll
  for (int nt = 0; nt < 2; ++nt) {
    int col = nt * 16 + r;
    float bias = nb[col];
#pragma unroll
    for (int i = 0; i < 8; ++i) {
      long e = r0 + half8 + i;
      int d = dst[e];
      atomicAdd(agg + (long)d * 32 + col, acc[nt][i] + bias);
    }
  }
}

// ---------------------------------------------------------------------------
// Node update: one node per wave32; row broadcast via __shfl.
// ---------------------------------------------------------------------------
template <int FIN>
__global__ __launch_bounds__(256)
void node_update_kernel(const float* __restrict__ xold,
                        const float* __restrict__ agg,
                        const float* __restrict__ lw,
                        const float* __restrict__ lb,
                        float* __restrict__ xnew,
                        _Float16* __restrict__ xh) {
  const int node = (int)((blockIdx.x * blockDim.x + threadIdx.x) >> 5);
  const int lane = threadIdx.x & 31;
  if (node >= N_NODES) return;
  const long base = (long)node * 32;
  float xv = (lane < FIN) ? xold[base + lane] : 0.f;
  float sum = agg[base + lane] + lb[lane];
#pragma unroll
  for (int k = 0; k < FIN; ++k)
    sum += __shfl(xv, k, 32) * lw[k * 32 + lane];
  sum = sum > 0.f ? sum : 0.f;
  xnew[base + lane] = sum;
  xh[base + lane] = (_Float16)sum;
}

// ---------------------------------------------------------------------------
__global__ __launch_bounds__(256)
void final_kernel(const float* __restrict__ x,
                  const float* __restrict__ lw,
                  const float* __restrict__ lb,
                  float* __restrict__ out) {
  const int node = (int)((blockIdx.x * blockDim.x + threadIdx.x) >> 5);
  const int lane = threadIdx.x & 31;
  if (node >= N_NODES) return;
  float p = x[(long)node * 32 + lane] * lw[lane];
#pragma unroll
  for (int d = 16; d > 0; d >>= 1) p += __shfl_down(p, d, 32);
  if (lane == 0) out[node] = p + lb[0];
}

// ---------------------------------------------------------------------------
extern "C" void kernel_launch(void* const* d_in, const int* in_sizes, int n_in,
                              void* d_out, int out_size, void* d_ws, size_t ws_size,
                              hipStream_t stream) {
  (void)in_sizes; (void)n_in; (void)out_size; (void)ws_size;

  const float* node_feature = (const float*)d_in[0];
  const float* edge_feature = (const float*)d_in[1];
  const int*   edge_index   = (const int*)d_in[2];
  const int* srcp = edge_index;
  const int* dstp = edge_index + N_EDGES;
  const float* n_w = (const float*)d_in[3];
  const float* n_b = (const float*)d_in[4];
  const float* e_w1 = (const float*)d_in[5];
  const float* e_b1 = (const float*)d_in[6];
  const float* e_w2 = (const float*)d_in[7];
  const float* e_b2 = (const float*)d_in[8];
  const float* e_w3 = (const float*)d_in[9];
  const float* e_b3 = (const float*)d_in[10];
  const float* netw[5], *netb[5], *linw[5], *linb[5];
  for (int i = 0; i < 5; ++i) {
    netw[i] = (const float*)d_in[11 + 4 * i];
    netb[i] = (const float*)d_in[12 + 4 * i];
    linw[i] = (const float*)d_in[13 + 4 * i];
    linb[i] = (const float*)d_in[14 + 4 * i];
  }
  const float* last_w = (const float*)d_in[31];
  const float* last_b = (const float*)d_in[32];

  // -------- workspace carve (256B aligned) --------
  char* ws = (char*)d_ws;
  size_t off = 0;
  auto carve = [&](size_t bytes) -> char* {
    char* p = ws + off;
    off += (bytes + 255) & ~(size_t)255;
    return p;
  };
  _Float16* w1p = (_Float16*)carve(W1_HALVES * sizeof(_Float16));
  _Float16* w2p = (_Float16*)carve(4 * 4 * 512 * sizeof(_Float16));
  _Float16* w3p = (_Float16*)carve(2 * 2 * 512 * sizeof(_Float16));
  _Float16* cnp[5];
  for (int i = 0; i < 5; ++i) cnp[i] = (_Float16*)carve(2 * 2 * 512 * sizeof(_Float16));
  _Float16* eh  = (_Float16*)carve((size_t)N_EDGES * 32 * sizeof(_Float16));
  _Float16* xh  = (_Float16*)carve((size_t)N_NODES * 32 * sizeof(_Float16));
  float* xA  = (float*)carve((size_t)N_NODES * 32 * sizeof(float));
  float* xB  = (float*)carve((size_t)N_NODES * 32 * sizeof(float));
  float* agg = (float*)carve((size_t)N_NODES * 32 * sizeof(float));

  // -------- weight packing (tiny) --------
  auto pack = [&](const float* W, _Float16* o, int K, int N, int KT) {
    int total = KT * (N >> 4) * 512;
    pack_b_kernel<<<(total + 255) / 256, 256, 0, stream>>>(W, o, K, N, KT);
  };
  pack(e_w1, w1p, 301, 128, 10);
  pack(e_w2, w2p, 128, 64, 4);
  pack(e_w3, w3p, 64, 32, 2);
  pack(netw[0], cnp[0], 48, 32, 2);
  for (int i = 1; i < 5; ++i) pack(netw[i], cnp[i], 64, 32, 2);

  // -------- node encoder --------
  node_encoder_kernel<<<(N_NODES + 255) / 256, 256, 0, stream>>>(
      node_feature, n_w, n_b, xA, xh);

  // -------- fused edge encoder (WMMA, TDM-staged W1) --------
  const int edgeBlocks = N_EDGES / (4 * 16);  // 15625
  const size_t encLds = (size_t)(W1_HALVES + 4 * 16 * 128 + 4 * 16 * 64) *
                        sizeof(_Float16);  // 106496 B
  edge_encoder_kernel<<<edgeBlocks, 128, encLds, stream>>>(
      edge_feature, w1p, e_b1, w2p, e_b2, w3p, e_b3, eh);

  // -------- 5 NNConv layers --------
  float* xcur = xA;
  float* xnxt = xB;
  const int nodeWaveBlocks = (N_NODES * 32 + 255) / 256;  // one wave per node
  for (int L = 0; L < 5; ++L) {
    (void)hipMemsetAsync(agg, 0, (size_t)N_NODES * 32 * sizeof(float), stream);
    if (L == 0) {
      nnconv_msg_kernel<16><<<edgeBlocks, 128, 0, stream>>>(
          xh, eh, srcp, dstp, cnp[0], netb[0], agg);
      node_update_kernel<16><<<nodeWaveBlocks, 256, 0, stream>>>(
          xcur, agg, linw[0], linb[0], xnxt, xh);
    } else {
      nnconv_msg_kernel<32><<<edgeBlocks, 128, 0, stream>>>(
          xh, eh, srcp, dstp, cnp[L], netb[L], agg);
      node_update_kernel<32><<<nodeWaveBlocks, 256, 0, stream>>>(
          xcur, agg, linw[L], linb[L], xnxt, xh);
    }
    float* tmp = xcur; xcur = xnxt; xnxt = tmp;
  }

  // -------- final head --------
  final_kernel<<<nodeWaveBlocks, 256, 0, stream>>>(
      xcur, last_w, last_b, (float*)d_out);
}
